// AttentionModel_39848706572853
// MI455X (gfx1250) — matrix operand order repaired and checked by reading, not compiled, not run on hardware
//
#include <hip/hip_runtime.h>

#define Bb 2
#define Ss 2048
#define Hh 16
#define Dd 64

typedef __attribute__((ext_vector_type(16))) _Float16 v16h;
typedef __attribute__((ext_vector_type(8)))  float    v8f;

static __device__ __forceinline__ v8f wmma_f16(v16h a, v16h b, v8f c) {
  // D = A(16x32 f16) * B(32x16 f16) + C(16x16 f32)
  return __builtin_amdgcn_wmma_f32_16x16x32_f16(
      /*neg_a=*/false, a, /*neg_b=*/false, b,
      /*c_mod=*/(short)0, c, /*reuse_a=*/false, /*reuse_b=*/false);
}

// native v_exp_f32 is 2^x; scores arrive pre-scaled by log2(e)
static __device__ __forceinline__ float fast_exp2(float x) {
#if __has_builtin(__builtin_amdgcn_exp2f)
  return __builtin_amdgcn_exp2f(x);
#else
  return exp2f(x);
#endif
}

// pair-pack f32 -> f16x2 (v_cvt_pk_rtz_f16_f32); use the builtin's own vector type
typedef decltype(__builtin_amdgcn_cvt_pkrtz(0.0f, 0.0f)) pkh2;

// 16 contiguous f16 (32B) -> full v16h fragment (B operands)
static __device__ __forceinline__ v16h ld16(const _Float16* p) {
  v16h r;
  ((uint4*)&r)[0] = ((const uint4*)p)[0];
  ((uint4*)&r)[1] = ((const uint4*)p)[1];
  return r;
}
// two 8-f16 (16B) chunks -> v16h fragment (A operands, contiguous k = 16*hi + s as called)
static __device__ __forceinline__ v16h ld8x2(const _Float16* p0, const _Float16* p1) {
  v16h r;
  ((uint4*)&r)[0] = *(const uint4*)p0;
  ((uint4*)&r)[1] = *(const uint4*)p1;
  return r;
}

// ---------------- Projection: out = x @ W^T + b (fp32), cast fp16 ----------------
// mode 0: Q  -> [B,H,S,D], scaled by log2(e)/sqrt(D)  (base-2 softmax folding)
// mode 1: K  -> [B,H,S,D]
// mode 2: V  -> [B,H,D,S]  (transposed so PV B-fragments load contiguously)
__global__ void __launch_bounds__(256) proj_kernel(
    const float* __restrict__ x, const float* __restrict__ W,
    const float* __restrict__ bias, _Float16* __restrict__ out, int mode)
{
  __shared__ float Wl[64][65];   // padded: bank-conflict-free column reads
  __shared__ float bl[64];
  __shared__ float xs[4][64];

  const int t = threadIdx.x;
#pragma unroll
  for (int i = 0; i < 16; ++i) {
    int li = t + i * 256;
    Wl[li >> 6][li & 63] = W[li];
  }
  if (t < 64) bl[t] = bias[t];

  const int rloc = t >> 6, d = t & 63;
  const long row = (long)blockIdx.x * 4 + rloc;   // row over [B,S,H]
  xs[rloc][d] = x[row * 64 + d];
  __syncthreads();

  float acc = bl[d];
#pragma unroll
  for (int k = 0; k < 64; ++k) acc += xs[rloc][k] * Wl[d][k];

  const int b   = (int)(row / (Ss * Hh));
  const int rem = (int)(row % (Ss * Hh));
  const int s = rem / Hh, h = rem % Hh;

  if (mode == 2) {
    out[(((long)b * Hh + h) * Dd + d) * Ss + s] = (_Float16)acc;
  } else {
    // Q: fold 1/sqrt(64) and log2(e) so the kernel can use native v_exp_f32 (2^x)
    const float QSCALE = 0.125f * 1.44269504088896340736f;
    float v = (mode == 0) ? acc * QSCALE : acc;
    out[(((long)b * Hh + h) * Ss + s) * Dd + d] = (_Float16)v;
  }
}

// ---------------- Flash attention: one wave per 16-row Q tile, Bc = 64 ----------------
// Scores computed TRANSPOSED (S^T = K_tile x Q^T) so each query's scores land in a
// single lane pair -> softmax reductions are in-register, one xor-16 exchange.
// Scores are pre-scaled by log2(e): probabilities via exp2 (native v_exp_f32).
__global__ void __launch_bounds__(256) attn_kernel(
    const _Float16* __restrict__ Qh,   // [B*H, S, D] fp16, pre-scaled
    const _Float16* __restrict__ Kh,   // [B*H, S, D] fp16
    const _Float16* __restrict__ Vt,   // [B*H, D, S] fp16 (transposed)
    float* __restrict__ out)           // [B*H, S, D] fp32
{
  __shared__ __align__(16) _Float16 Pl[8][16][64];  // per-wave P staging [query][key]

  const int tid  = threadIdx.x;
  const int w    = tid >> 5;
  const int lane = tid & 31;
  const int l    = lane & 15;          // query column (S^T) / d column (PV out)
  const int hi   = lane >> 4;          // lane half

  const int bh    = blockIdx.x >> 4;                 // S/128 = 16 q-tiles per (b,h)
  const int qbase = ((blockIdx.x & 15) << 7) + (w << 4);

  const _Float16* Qbh = Qh + (size_t)bh * Ss * Dd;
  const _Float16* Kbh = Kh + (size_t)bh * Ss * Dd;
  const _Float16* Vbh = Vt + (size_t)bh * Dd * Ss;

  // B fragments of Q^T (d x 16 queries), resident for the whole loop.
  const _Float16* Qr = Qbh + (size_t)(qbase + l) * Dd;
  const v16h Bq0 = ld16(Qr +      hi * 16);   // d 0..31
  const v16h Bq1 = ld16(Qr + 32 + hi * 16);   // d 32..63

  v8f acc0 = {}, acc1 = {}, acc2 = {}, acc3 = {};
  float mrun = -3.0e38f, lrun = 0.0f;          // scalar stats for query l (lane pair)

  for (int kb = 0; kb < Ss; kb += 64) {
    // ---- S^T tiles: st[t] = K(16 keys x 64 d) * Q^T(64 d x 16 q) ----
    v8f st[4];
#pragma unroll
    for (int t4 = 0; t4 < 4; ++t4) {
      const _Float16* Kr = Kbh + (size_t)(kb + t4 * 16 + l) * Dd;
      v16h Ak0 = ld8x2(Kr +      hi * 16, Kr +  8 + hi * 16);   // d 0..31
      v16h Ak1 = ld8x2(Kr + 32 + hi * 16, Kr + 40 + hi * 16);   // d 32..63
      v8f z = {};
      st[t4] = wmma_f16(Ak0, Bq0, z);
      st[t4] = wmma_f16(Ak1, Bq1, st[t4]);
    }
    if (kb + 64 < Ss) __builtin_prefetch(Kbh + (size_t)(kb + 64 + l) * Dd, 0, 0);

    // ---- softmax stats for query l: in-lane tree + one lane-pair exchange ----
    float tm = st[0][0];
#pragma unroll
    for (int t4 = 0; t4 < 4; ++t4)
#pragma unroll
      for (int r = 0; r < 8; ++r)
        tm = fmaxf(tm, st[t4][r]);
    tm = fmaxf(tm, __shfl_xor(tm, 16, 32));
    const float mn    = fmaxf(mrun, tm);
    const float alpha = fast_exp2(mrun - mn);   // base-2: native v_exp_f32
    mrun = mn;

    // broadcast alpha into the PV accumulator's row (=query) layout, rescale O
    float ar[8];
#pragma unroll
    for (int r = 0; r < 8; ++r) ar[r] = __shfl(alpha, hi * 8 + r, 32);
#pragma unroll
    for (int r = 0; r < 8; ++r) {
      acc0[r] *= ar[r]; acc1[r] *= ar[r];
      acc2[r] *= ar[r]; acc3[r] *= ar[r];
    }

    // ---- exp2, pack pairs (v_cvt_pk_rtz), one b128 LDS store per tile ----
    float rs = 0.0f;
#pragma unroll
    for (int t4 = 0; t4 < 4; ++t4) {
      float p[8];
#pragma unroll
      for (int r = 0; r < 8; ++r) {
        p[r] = fast_exp2(st[t4][r] - mn);
        rs += p[r];
      }
      union { uint4 u; pkh2 h2[4]; } pk;
#pragma unroll
      for (int r = 0; r < 4; ++r)
        pk.h2[r] = __builtin_amdgcn_cvt_pkrtz(p[2 * r], p[2 * r + 1]);
      *(uint4*)&Pl[w][l][t4 * 16 + hi * 8] = pk.u;
    }
    rs += __shfl_xor(rs, 16, 32);
    lrun = lrun * alpha + rs;

    asm volatile("s_wait_dscnt 0x0" ::: "memory");   // DS in-order per wave

    // reload P as A fragments (16 queries x 64 keys = two K=32 frags)
    const _Float16* Pr = &Pl[w][l][0];
    const v16h Ap0 = ld8x2(Pr +      hi * 16, Pr +  8 + hi * 16);  // keys 0..31
    const v16h Ap1 = ld8x2(Pr + 32 + hi * 16, Pr + 40 + hi * 16);  // keys 32..63

    // ---- O += P(16x64) * V(64x64): B-fragments contiguous from transposed V ----
    acc0 = wmma_f16(Ap0, ld16(Vbh + (size_t)( 0 + l) * Ss + kb +      hi * 16), acc0);
    acc0 = wmma_f16(Ap1, ld16(Vbh + (size_t)( 0 + l) * Ss + kb + 32 + hi * 16), acc0);
    acc1 = wmma_f16(Ap0, ld16(Vbh + (size_t)(16 + l) * Ss + kb +      hi * 16), acc1);
    acc1 = wmma_f16(Ap1, ld16(Vbh + (size_t)(16 + l) * Ss + kb + 32 + hi * 16), acc1);
    acc2 = wmma_f16(Ap0, ld16(Vbh + (size_t)(32 + l) * Ss + kb +      hi * 16), acc2);
    acc2 = wmma_f16(Ap1, ld16(Vbh + (size_t)(32 + l) * Ss + kb + 32 + hi * 16), acc2);
    acc3 = wmma_f16(Ap0, ld16(Vbh + (size_t)(48 + l) * Ss + kb +      hi * 16), acc3);
    acc3 = wmma_f16(Ap1, ld16(Vbh + (size_t)(48 + l) * Ss + kb + 32 + hi * 16), acc3);
  }

  // ---- epilogue: broadcast 1/l into accumulator row layout, write O ----
  float linv[8];
#pragma unroll
  for (int r = 0; r < 8; ++r) linv[r] = 1.0f / __shfl(lrun, hi * 8 + r, 32);
#pragma unroll
  for (int r = 0; r < 8; ++r) {
    const int row = hi * 8 + r;
    const size_t o = ((size_t)bh * Ss + qbase + row) * Dd + l;
    out[o +  0] = acc0[r] * linv[r];
    out[o + 16] = acc1[r] * linv[r];
    out[o + 32] = acc2[r] * linv[r];
    out[o + 48] = acc3[r] * linv[r];
  }
}

extern "C" void kernel_launch(void* const* d_in, const int* in_sizes, int n_in,
                              void* d_out, int out_size, void* d_ws, size_t ws_size,
                              hipStream_t stream) {
  const float* q  = (const float*)d_in[0];
  const float* k  = (const float*)d_in[1];
  const float* v  = (const float*)d_in[2];
  const float* Wq = (const float*)d_in[3];
  const float* bq = (const float*)d_in[4];
  const float* Wk = (const float*)d_in[5];
  const float* bk = (const float*)d_in[6];
  const float* Wv = (const float*)d_in[7];
  const float* bv = (const float*)d_in[8];

  const size_t N = (size_t)Bb * Hh * Ss * Dd;   // 4,194,304 elements
  _Float16* Qh = (_Float16*)d_ws;               //  8 MB
  _Float16* Kh = Qh + N;                        //  8 MB
  _Float16* Vt = Kh + N;                        //  8 MB

  const int projBlocks = (Bb * Ss * Hh) / 4;    // 16384
  proj_kernel<<<projBlocks, 256, 0, stream>>>(q, Wq, bq, Qh, 0);
  proj_kernel<<<projBlocks, 256, 0, stream>>>(k, Wk, bk, Kh, 1);
  proj_kernel<<<projBlocks, 256, 0, stream>>>(v, Wv, bv, Vt, 2);

  const int attnBlocks = Bb * Hh * (Ss / 128);  // 512 blocks x 8 waves
  attn_kernel<<<attnBlocks, 256, 0, stream>>>(Qh, Kh, Vt, (float*)d_out);

  (void)in_sizes; (void)n_in; (void)out_size; (void)ws_size;
}
